// GAT_45792941309972
// MI455X (gfx1250) — matrix operand order repaired and checked by reading, not compiled, not run on hardware
//
#include <hip/hip_runtime.h>
#include <hip/hip_bf16.h>
#include <math.h>

#define NEG_SLOPE 0.2f

typedef __attribute__((ext_vector_type(16))) _Float16 v16h;
typedef __attribute__((ext_vector_type(8)))  _Float16 half8;
typedef __attribute__((ext_vector_type(8)))  float    v8f;

__device__ __forceinline__ float leaky(float v) {
    return v > 0.f ? v : NEG_SLOPE * v;
}

// float atomic max via signed/unsigned int atomics (works with -inf init)
__device__ __forceinline__ void atomicMaxF(float* addr, float value) {
    if (value >= 0.f)
        atomicMax((int*)addr, __float_as_int(value));
    else
        atomicMin((unsigned int*)addr, __float_as_uint(value));
}

__global__ void fill_f32(float* __restrict__ p, float v, long long n) {
    long long i = (long long)blockIdx.x * blockDim.x + threadIdx.x;
    if (i < n) p[i] = v;
}

__global__ void f32_to_f16_kernel(const float* __restrict__ in, _Float16* __restrict__ out,
                                  long long n) {
    long long i = (long long)blockIdx.x * blockDim.x + threadIdx.x;
    if (i < n) out[i] = (_Float16)in[i];
}

// Pack B[K,64] (fp32) into WMMA f16 fragment layout:
//   Bp[((kt*4 + tn)*32 + lane)*16 + j] = B[kt*32 + (j>>3)*16 + (lane>>4)*8 + (j&7)][tn*16 + (lane&15)]
__global__ void pack_b_kernel(const float* __restrict__ B, _Float16* __restrict__ Bp, int K) {
    int i = blockIdx.x * blockDim.x + threadIdx.x;
    int total = (K >> 5) * 4 * 32 * 16;
    if (i >= total) return;
    int j    = i & 15;
    int lane = (i >> 4) & 31;
    int t    = (i >> 9) & 3;
    int kt   = i >> 11;
    int m = lane & 15, kh = lane >> 4;
    int krow = (kt << 5) + ((j >> 3) << 4) + (kh << 3) + (j & 7);
    Bp[i] = (_Float16)B[(size_t)krow * 64 + t * 16 + m];
}

// C[M,64] = A[M,K](f16) @ B[K,64](fragment-packed f16), f32 accum.
// One wave per 16-row stripe: 4 WMMA tiles across the 64 columns.
// Requires M % 16 == 0, K % 32 == 0. EXEC all-ones at every WMMA.
__global__ void __launch_bounds__(256) gemm16x64_wmma(
    const _Float16* __restrict__ A, const _Float16* __restrict__ Bp,
    float* __restrict__ C, int M, int K) {
    int wave = (blockIdx.x * blockDim.x + threadIdx.x) >> 5;
    int lane = threadIdx.x & 31;
    int tiles_m = M >> 4;
    if (wave >= tiles_m) return;          // wave-uniform
    int row0 = wave << 4;
    int m = lane & 15, kh = lane >> 4;
    int ktiles = K >> 5;
    const _Float16* arow = A + (size_t)(row0 + m) * K;
    v8f acc[4] = {{}, {}, {}, {}};
    for (int kt = 0; kt < ktiles; ++kt) {
        int k0 = kt << 5;
        // A 16x32 fragment: two contiguous 8-half runs per lane (b128 loads)
        half8 lo = *(const half8*)(arow + k0 + (kh << 3));
        half8 hi = *(const half8*)(arow + k0 + 16 + (kh << 3));
        v16h a = __builtin_shufflevector(lo, hi, 0, 1, 2, 3, 4, 5, 6, 7,
                                         8, 9, 10, 11, 12, 13, 14, 15);
#pragma unroll
        for (int t = 0; t < 4; ++t) {
            v16h b = *(const v16h*)(Bp + (((size_t)kt * 4 + t) * 32 + lane) * 16);
            acc[t] = __builtin_amdgcn_wmma_f32_16x16x32_f16(false, a, false, b,
                                                            (short)0, acc[t], false, false);
        }
    }
#pragma unroll
    for (int t = 0; t < 4; ++t) {
#pragma unroll
        for (int r = 0; r < 8; ++r) {
            int crow = row0 + r + (kh << 3);
            C[(size_t)crow * 64 + t * 16 + m] = acc[t][r];
        }
    }
}

// a_src/a_dst for layer 1: per (node, head) dot of 8 channels
__global__ void att1_kernel(const float* __restrict__ h1,
                            const float* __restrict__ att_src,
                            const float* __restrict__ att_dst,
                            float* __restrict__ a_src, float* __restrict__ a_dst,
                            int n) {
    int i = blockIdx.x * blockDim.x + threadIdx.x;
    if (i >= n * 8) return;
    int node = i >> 3, h = i & 7;
    const float* hp = h1 + (size_t)node * 64 + h * 8;
    float s = 0.f, d = 0.f;
#pragma unroll
    for (int c = 0; c < 8; ++c) {
        float v = hp[c];
        s += v * att_src[h * 8 + c];
        d += v * att_dst[h * 8 + c];
    }
    a_src[i] = s; a_dst[i] = d;
}

// a_src/a_dst for layer 2: per node dot of 64 channels (1 head)
__global__ void att2_kernel(const float* __restrict__ h2,
                            const float* __restrict__ att_src,
                            const float* __restrict__ att_dst,
                            float* __restrict__ a_src, float* __restrict__ a_dst,
                            int n) {
    int i = blockIdx.x * blockDim.x + threadIdx.x;
    if (i >= n) return;
    const float* hp = h2 + (size_t)i * 64;
    float s = 0.f, d = 0.f;
    for (int c = 0; c < 64; ++c) {
        float v = hp[c];
        s += v * att_src[c];
        d += v * att_dst[c];
    }
    a_src[i] = s; a_dst[i] = d;
}

// pass 1: segment max of leaky(logit) per destination
__global__ void edge_max_kernel(const int* __restrict__ src, const int* __restrict__ dst,
                                const float* __restrict__ a_src, const float* __restrict__ a_dst,
                                float* __restrict__ maxb, long long EH, int H) {
    long long i = (long long)blockIdx.x * blockDim.x + threadIdx.x;
    if (i >= EH) return;
    int e = (int)(i / H), h = (int)(i % H);
    int s = src[e], d = dst[e];
    float v = leaky(a_src[(size_t)s * H + h] + a_dst[(size_t)d * H + h]);
    atomicMaxF(&maxb[(size_t)d * H + h], v);
}

// pass 2: segment sum of exp(logit - max)
__global__ void edge_denom_kernel(const int* __restrict__ src, const int* __restrict__ dst,
                                  const float* __restrict__ a_src, const float* __restrict__ a_dst,
                                  const float* __restrict__ maxb, float* __restrict__ denom,
                                  long long EH, int H) {
    long long i = (long long)blockIdx.x * blockDim.x + threadIdx.x;
    if (i >= EH) return;
    int e = (int)(i / H), h = (int)(i % H);
    int s = src[e], d = dst[e];
    float v = leaky(a_src[(size_t)s * H + h] + a_dst[(size_t)d * H + h]);
    float ex = expf(v - maxb[(size_t)d * H + h]);
    atomicAdd(&denom[(size_t)d * H + h], ex);
}

// pass 3: out[dst] += alpha * h[src]  (C channels per head)
__global__ void edge_msg_kernel(const int* __restrict__ src, const int* __restrict__ dst,
                                const float* __restrict__ a_src, const float* __restrict__ a_dst,
                                const float* __restrict__ maxb, const float* __restrict__ denom,
                                const float* __restrict__ hfeat, float* __restrict__ out,
                                long long EH, int H, int C) {
    long long i = (long long)blockIdx.x * blockDim.x + threadIdx.x;
    if (i >= EH) return;
    int e = (int)(i / H), h = (int)(i % H);
    int s = src[e], d = dst[e];
    float v = leaky(a_src[(size_t)s * H + h] + a_dst[(size_t)d * H + h]);
    float alpha = expf(v - maxb[(size_t)d * H + h]) / denom[(size_t)d * H + h];
    const float* hp = hfeat + ((size_t)s * H + h) * C;
    float* op = out + ((size_t)d * H + h) * C;
    for (int c = 0; c < C; ++c)
        atomicAdd(&op[c], hp[c] * alpha);
}

// x1h = (f16) elu(out1 + b1)  -- f16 output feeds the layer-2 WMMA GEMM directly
__global__ void elu_bias_f16_kernel(const float* __restrict__ in, const float* __restrict__ bias,
                                    _Float16* __restrict__ out, long long n, int F) {
    long long i = (long long)blockIdx.x * blockDim.x + threadIdx.x;
    if (i >= n) return;
    float v = in[i] + bias[(int)(i % F)];
    out[i] = (_Float16)(v > 0.f ? v : expm1f(v));
}

// bias + log_softmax + argmax + masked NLL accumulation; writes pred/label as float
__global__ void finalize_kernel(const float* __restrict__ out2, const float* __restrict__ b2,
                                const unsigned char* __restrict__ mask,
                                const int* __restrict__ label,
                                float* __restrict__ dout, float* __restrict__ accum, int n) {
    int i = blockIdx.x * blockDim.x + threadIdx.x;
    if (i >= n) return;
    const float* row = out2 + (size_t)i * 64;
    float mx = -INFINITY; int arg = 0;
    for (int c = 0; c < 64; ++c) {
        float v = row[c] + b2[c];
        if (v > mx) { mx = v; arg = c; }
    }
    float sum = 0.f;
    for (int c = 0; c < 64; ++c)
        sum += expf(row[c] + b2[c] - mx);
    int lb = label[i];
    float logp = (row[lb] + b2[lb]) - mx - logf(sum);
    float m = mask[i] ? 1.f : 0.f;
    atomicAdd(&accum[0], -logp * m);
    atomicAdd(&accum[1], m);
    dout[1 + i]     = (float)arg;
    dout[1 + n + i] = (float)lb;
}

__global__ void loss_kernel(const float* __restrict__ accum, float* __restrict__ dout) {
    dout[0] = accum[0] / accum[1];
}

static inline int cdiv_ll(long long a, long long b) { return (int)((a + b - 1) / b); }

extern "C" void kernel_launch(void* const* d_in, const int* in_sizes, int n_in,
                              void* d_out, int out_size, void* d_ws, size_t ws_size,
                              hipStream_t stream) {
    const float* feat = (const float*)d_in[1];
    const int*   ei   = (const int*)d_in[2];
    const unsigned char* mask = (const unsigned char*)d_in[3];
    const int*   label = (const int*)d_in[4];
    const float* W1  = (const float*)d_in[5];
    const float* as1 = (const float*)d_in[6];
    const float* ad1 = (const float*)d_in[7];
    const float* b1  = (const float*)d_in[8];
    const float* W2  = (const float*)d_in[9];
    const float* as2 = (const float*)d_in[10];
    const float* ad2 = (const float*)d_in[11];
    const float* b2  = (const float*)d_in[12];

    const int n    = in_sizes[0];            // N = 100000 (multiple of 16)
    const int IN_F = in_sizes[1] / n;        // 128
    const int E    = in_sizes[2] / 2;        // 1,700,000
    const int* src = ei;
    const int* dst = ei + E;

    // ---- workspace layout (float offsets; all multiples of 8 -> 32B aligned) ----
    float* ws    = (float*)d_ws;
    float* bufA  = ws;                        // [n*64] h1, then h2 (f32)
    float* bufB  = bufA + (size_t)n * 64;     // [n*64] out1, then out2 (f32)
    float* a_s   = bufB + (size_t)n * 64;     // [n*8]
    float* a_d   = a_s  + (size_t)n * 8;      // [n*8]
    float* mxb   = a_d  + (size_t)n * 8;      // [n*8]
    float* dnm   = mxb  + (size_t)n * 8;      // [n*8]
    float* accum = dnm  + (size_t)n * 8;      // [8] (2 used)
    _Float16* A16h = (_Float16*)(accum + 8);          // [n*128] f16 feat
    _Float16* x1h  = A16h + (size_t)n * IN_F;         // [n*64]  f16 elu output
    _Float16* Bp1  = x1h + (size_t)n * 64;            // [8192]  packed W1 fragments
    _Float16* Bp2  = Bp1 + 8192;                      // [4096]  packed W2 fragments

    float* dout = (float*)d_out;
    const int BT = 256;

    // ---- one-time (per call) conversions / packs ----
    f32_to_f16_kernel<<<cdiv_ll((long long)n * IN_F, BT), BT, 0, stream>>>(
        feat, A16h, (long long)n * IN_F);
    pack_b_kernel<<<cdiv_ll((IN_F / 32) * 2048, BT), BT, 0, stream>>>(W1, Bp1, IN_F);
    pack_b_kernel<<<cdiv_ll((64 / 32) * 2048, BT), BT, 0, stream>>>(W2, Bp2, 64);

    // ---------------- Layer 1: h1 = feat @ W1 (WMMA) ----------------
    {
        long long threads = (long long)(n >> 4) * 32;
        gemm16x64_wmma<<<cdiv_ll(threads, BT), BT, 0, stream>>>(A16h, Bp1, bufA, n, IN_F);
    }
    att1_kernel<<<cdiv_ll((long long)n * 8, BT), BT, 0, stream>>>(bufA, as1, ad1, a_s, a_d, n);

    fill_f32<<<cdiv_ll((long long)n * 8, BT), BT, 0, stream>>>(mxb, -INFINITY, (long long)n * 8);
    fill_f32<<<cdiv_ll((long long)n * 8, BT), BT, 0, stream>>>(dnm, 0.f, (long long)n * 8);
    fill_f32<<<cdiv_ll((long long)n * 64, BT), BT, 0, stream>>>(bufB, 0.f, (long long)n * 64);
    fill_f32<<<1, BT, 0, stream>>>(accum, 0.f, 8);

    {
        long long EH = (long long)E * 8;
        edge_max_kernel<<<cdiv_ll(EH, BT), BT, 0, stream>>>(src, dst, a_s, a_d, mxb, EH, 8);
        edge_denom_kernel<<<cdiv_ll(EH, BT), BT, 0, stream>>>(src, dst, a_s, a_d, mxb, dnm, EH, 8);
        edge_msg_kernel<<<cdiv_ll(EH, BT), BT, 0, stream>>>(src, dst, a_s, a_d, mxb, dnm,
                                                            bufA, bufB, EH, 8, 8);
    }
    elu_bias_f16_kernel<<<cdiv_ll((long long)n * 64, BT), BT, 0, stream>>>(
        bufB, b1, x1h, (long long)n * 64, 64);

    // ---------------- Layer 2: h2 = x1 @ W2 (WMMA) ----------------
    {
        long long threads = (long long)(n >> 4) * 32;
        gemm16x64_wmma<<<cdiv_ll(threads, BT), BT, 0, stream>>>(x1h, Bp2, bufA, n, 64);
    }
    att2_kernel<<<cdiv_ll((long long)n, BT), BT, 0, stream>>>(bufA, as2, ad2, a_s, a_d, n);

    fill_f32<<<cdiv_ll((long long)n, BT), BT, 0, stream>>>(mxb, -INFINITY, (long long)n);
    fill_f32<<<cdiv_ll((long long)n, BT), BT, 0, stream>>>(dnm, 0.f, (long long)n);
    fill_f32<<<cdiv_ll((long long)n * 64, BT), BT, 0, stream>>>(bufB, 0.f, (long long)n * 64);

    {
        long long EH = (long long)E;
        edge_max_kernel<<<cdiv_ll(EH, BT), BT, 0, stream>>>(src, dst, a_s, a_d, mxb, EH, 1);
        edge_denom_kernel<<<cdiv_ll(EH, BT), BT, 0, stream>>>(src, dst, a_s, a_d, mxb, dnm, EH, 1);
        edge_msg_kernel<<<cdiv_ll(EH, BT), BT, 0, stream>>>(src, dst, a_s, a_d, mxb, dnm,
                                                            bufA, bufB, EH, 1, 64);
    }

    finalize_kernel<<<cdiv_ll((long long)n, BT), BT, 0, stream>>>(
        bufB, b2, mask, label, dout, accum, n);
    loss_kernel<<<1, 1, 0, stream>>>(accum, dout);
}
